// RotaryPEMultiHeadAttention_9818295239475
// MI455X (gfx1250) — matrix operand order repaired and checked by reading, not compiled, not run on hardware
//
#include <hip/hip_runtime.h>
#include <hip/hip_bf16.h>

typedef __attribute__((ext_vector_type(16))) _Float16 v16h;
typedef __attribute__((ext_vector_type(8)))  _Float16 v8h;
typedef __attribute__((ext_vector_type(4)))  _Float16 v4h;
typedef __attribute__((ext_vector_type(8)))  float    v8f;

union AV { v16h v; v8h h[2]; };

#define B_   4
#define S_   2048
#define H_   16
#define DK_  64
#define HID_ 1024
#define LSTR 40                 // LDS B-tile row stride in halfs (80B)
#define BUFH (64 * LSTR)        // halfs per LDS buffer
#define BUFB (BUFH * 2)         // bytes per LDS buffer

__device__ __forceinline__ v8f wmma_f16(v16h a, v16h b, v8f c) {
  return __builtin_amdgcn_wmma_f32_16x16x32_f16(false, a, false, b, (short)0, c,
                                                false, false);
}

__device__ __forceinline__ void async_ld_b128(uint32_t lds_byte, const void* gaddr) {
  // GLOBAL_LOAD_ASYNC_TO_LDS_B128 (GV mode): copies 16B from global memory into LDS, tracked by ASYNCcnt
  asm volatile("global_load_async_to_lds_b128 %0, %1, off"
               :: "v"(lds_byte), "v"(gaddr) : "memory");
}

// ---------------- f32 -> f16 convert (x) ----------------
__global__ void k_cvt_x(const float* __restrict__ x, _Float16* __restrict__ xh, int n) {
  int i = (blockIdx.x * blockDim.x + threadIdx.x) * 4;
  if (i < n) {
    float4 f = *(const float4*)(x + i);
    v4h o;
    o.x = (_Float16)f.x; o.y = (_Float16)f.y;
    o.z = (_Float16)f.z; o.w = (_Float16)f.w;
    *(v4h*)(xh + i) = o;
  }
}

// ---------------- transpose + convert weight: Wt[n][k] = W[k][n] ----------------
__global__ void k_cvt_wt(const float* __restrict__ W, _Float16* __restrict__ Wt) {
  int idx = blockIdx.x * blockDim.x + threadIdx.x;   // over 1024*1024
  int nn = idx >> 10;
  int kk = idx & 1023;
  Wt[idx] = (_Float16)W[kk * HID_ + nn];
}

// ---------------- fused QKV projection + bias + RoPE ----------------
// 4 waves / block; block tile = 128 rows x 64 cols (one head); each wave owns
// 2 M-subtiles (32 rows). Weight tile async-staged in LDS, double buffered with
// data-driven buffer select (single loop body, in-place WMMA accumulation).
// mode 0/1: Q/K (RoPE, out [B][H][S][64]); mode 2: V (out transposed [B][H][64][S])
__global__ __launch_bounds__(128)
void k_qkv(const _Float16* __restrict__ xh, const _Float16* __restrict__ Wt,
           const float* __restrict__ bias, _Float16* __restrict__ out, int mode) {
  __shared__ __align__(16) _Float16 lbuf[2][BUFH];

  const int tid  = threadIdx.x;
  const int lane = tid & 31;
  const int wv   = tid >> 5;           // wave 0..3
  const int n  = lane & 15;
  const int hi = lane >> 4;

  const int mt = blockIdx.x;           // 0..63 : 128-row tiles over B*S
  const int h  = blockIdx.y;           // head
  const int b  = mt >> 4;              // 16 tiles of 128 rows per batch
  const int tb = (mt & 15) * 128 + wv * 32;  // wave's first row (seq position)
  const int col0 = h * DK_;

  // async-copy assignment: wave wv copies B-tile rows [wv*16, wv*16+16)
  const int cr0 = wv * 16 + ((lane >> 2) & 7);  // row for q=0 (q=1 adds 8)
  const int cb  = (lane & 3) * 16;              // byte offset within 64B row

  const _Float16* xrow0 = xh + (size_t)(b * S_ + tb + n) * HID_;
  const _Float16* xrow1 = xrow0 + (size_t)16 * HID_;
  const char* wsrc = (const char*)(Wt + (size_t)(col0 + cr0) * HID_) + cb;
  const uint32_t lds0 =
      (uint32_t)(uintptr_t)&lbuf[0][0] + (uint32_t)cr0 * (LSTR * 2) + cb;

  auto issue = [&](uint32_t dst, int kb) {
    const char* g0 = wsrc + (size_t)kb * 2;
    async_ld_b128(dst, g0);
    async_ld_b128(dst + 8 * (LSTR * 2), g0 + 8 * (HID_ * 2));
  };

  v8f acc[2][4] = {};

  auto compute = [&](const _Float16* lb, int kb) {
    AV a0, a1;
    a0.h[0] = *(const v8h*)(xrow0 + kb + 8 * hi);
    a0.h[1] = *(const v8h*)(xrow0 + kb + 16 + 8 * hi);
    a1.h[0] = *(const v8h*)(xrow1 + kb + 8 * hi);
    a1.h[1] = *(const v8h*)(xrow1 + kb + 16 + 8 * hi);
    AV bv[4];
#pragma unroll
    for (int dt = 0; dt < 4; ++dt) {
      const _Float16* bp = lb + (dt * 16 + n) * LSTR + 8 * hi;
      bv[dt].h[0] = *(const v8h*)(bp);
      bv[dt].h[1] = *(const v8h*)(bp + 16);
    }
#pragma unroll
    for (int dt = 0; dt < 4; ++dt) {
      acc[0][dt] = wmma_f16(a0.v, bv[dt].v, acc[0][dt]);
      acc[1][dt] = wmma_f16(a1.v, bv[dt].v, acc[1][dt]);
    }
  };

  issue(lds0, 0);
#pragma unroll 1
  for (int i = 0; i < 31; ++i) {                 // single-body steady state
    issue(lds0 + (uint32_t)(((i & 1) ^ 1) * BUFB), (i + 1) * 32);
    asm volatile("s_wait_asynccnt 2" ::: "memory");
    __syncthreads();                             // buf[i&1] fully populated
    compute(&lbuf[0][0] + (i & 1) * BUFH, i * 32);
    __syncthreads();                             // safe to refill buf[i&1]
  }
  asm volatile("s_wait_asynccnt 0" ::: "memory"); // peeled last iteration
  __syncthreads();
  compute(&lbuf[1][0], 31 * 32);

  float bc[4];
#pragma unroll
  for (int dt = 0; dt < 4; ++dt) bc[dt] = bias[col0 + dt * 16 + n];

#pragma unroll
  for (int mi = 0; mi < 2; ++mi) {
    const int ts = tb + mi * 16;
#pragma unroll
    for (int dt = 0; dt < 4; ++dt)
#pragma unroll
      for (int r = 0; r < 8; ++r) acc[mi][dt][r] += bc[dt];

    if (mode < 2) {
      // RoPE: pairs (d, d+16) for d<16 are (acc0, acc1) at identical (lane, r).
      float theta = __expf(-0.5756462732485115f * (float)n);  // 10000^(-n/16)
#pragma unroll
      for (int r = 0; r < 8; ++r) {
        float t = (float)(ts + r + 8 * hi);
        float sv, cv;
        __sincosf(t * theta, &sv, &cv);
        float v0 = acc[mi][0][r], v1 = acc[mi][1][r];
        acc[mi][0][r] = v0 * cv - v1 * sv;
        acc[mi][1][r] = v1 * cv + v0 * sv;
      }
    }

    if (mode == 2) {
      _Float16* base = out + (size_t)(b * H_ + h) * DK_ * S_;   // [B][H][64][S]
#pragma unroll
      for (int dt = 0; dt < 4; ++dt)
#pragma unroll
        for (int r = 0; r < 8; ++r)
          base[(size_t)(dt * 16 + n) * S_ + ts + r + 8 * hi] = (_Float16)acc[mi][dt][r];
    } else {
      _Float16* base = out + ((size_t)(b * H_ + h) * S_ + ts) * DK_;  // [B][H][S][64]
#pragma unroll
      for (int dt = 0; dt < 4; ++dt)
#pragma unroll
        for (int r = 0; r < 8; ++r)
          base[(size_t)(r + 8 * hi) * DK_ + dt * 16 + n] = (_Float16)acc[mi][dt][r];
    }
  }
}

// ---------------- flash attention (non-causal), one wave per (b,h,qtile) ----------------
__global__ __launch_bounds__(32)
void k_attn(const _Float16* __restrict__ Qh, const _Float16* __restrict__ Kh,
            const _Float16* __restrict__ Vt, _Float16* __restrict__ ctx) {
  __shared__ __align__(16) float    s_sc[16][32];
  __shared__ __align__(16) _Float16 s_p[16][32];
  __shared__ float s_alpha[16];
  __shared__ float s_linv[16];

  const int lane = threadIdx.x;
  const int n  = lane & 15;
  const int hi = lane >> 4;
  const int bx = blockIdx.x;
  const int qt = bx & 127;
  const int h  = (bx >> 7) & 15;
  const int b  = bx >> 11;
  const int t0 = qt << 4;
  const size_t bh = (size_t)(b * H_ + h);

  const _Float16* Qrow = Qh + (bh * S_ + t0 + n) * DK_;
  AV qa0, qa1;
  qa0.h[0] = *(const v8h*)(Qrow + 16 * hi);
  qa0.h[1] = *(const v8h*)(Qrow + 8 + 16 * hi);
  qa1.h[0] = *(const v8h*)(Qrow + 32 + 16 * hi);
  qa1.h[1] = *(const v8h*)(Qrow + 40 + 16 * hi);

  v8f acc[4] = {};
  float mrun = -__builtin_inff();
  float lrun = 0.f;
  const float scale = 0.125f;  // 1/sqrt(64)

  for (int j = 0; j < S_; j += 32) {
    const _Float16* K0 = Kh + (bh * S_ + j + n) * DK_;
    const _Float16* K1 = K0 + 16 * DK_;
    v16h kb0lo = *(const v16h*)(K0 + 16 * hi);
    v16h kb0hi = *(const v16h*)(K0 + 32 + 16 * hi);
    v16h kb1lo = *(const v16h*)(K1 + 16 * hi);
    v16h kb1hi = *(const v16h*)(K1 + 32 + 16 * hi);
    v8f c0 = {}, c1 = {};
    c0 = wmma_f16(qa0.v, kb0lo, c0);
    c0 = wmma_f16(qa1.v, kb0hi, c0);
    c1 = wmma_f16(qa0.v, kb1lo, c1);
    c1 = wmma_f16(qa1.v, kb1hi, c1);

#pragma unroll
    for (int r = 0; r < 8; ++r) {
      s_sc[r + 8 * hi][n]      = c0[r] * scale;
      s_sc[r + 8 * hi][16 + n] = c1[r] * scale;
    }
    __syncthreads();

    if (lane < 16) {
      const int m = lane;
      float tmax = s_sc[m][0];
      for (int jj = 1; jj < 32; ++jj) tmax = fmaxf(tmax, s_sc[m][jj]);
      float nmax  = fmaxf(mrun, tmax);
      float alpha = __expf(mrun - nmax);
      float sum = 0.f;
      for (int jj = 0; jj < 32; ++jj) {
        float p = __expf(s_sc[m][jj] - nmax);
        s_p[m][jj] = (_Float16)p;
        sum += p;
      }
      lrun = lrun * alpha + sum;
      mrun = nmax;
      s_alpha[m] = alpha;
    }
    __syncthreads();

    AV pa;
    pa.h[0] = *(const v8h*)(&s_p[n][16 * hi]);
    pa.h[1] = *(const v8h*)(&s_p[n][8 + 16 * hi]);

    float al[8];
#pragma unroll
    for (int r = 0; r < 8; ++r) al[r] = s_alpha[r + 8 * hi];
#pragma unroll
    for (int dt = 0; dt < 4; ++dt)
#pragma unroll
      for (int r = 0; r < 8; ++r) acc[dt][r] *= al[r];

#pragma unroll
    for (int dt = 0; dt < 4; ++dt) {
      const _Float16* Vrow = Vt + (bh * DK_ + dt * 16 + n) * S_;
      v16h vb = *(const v16h*)(Vrow + j + 16 * hi);
      acc[dt] = wmma_f16(pa.v, vb, acc[dt]);
    }
    __syncthreads();
  }

  if (lane < 16) s_linv[lane] = 1.f / lrun;
  __syncthreads();

  _Float16* crow = ctx + ((size_t)b * S_ + t0) * HID_ + h * DK_;
#pragma unroll
  for (int dt = 0; dt < 4; ++dt)
#pragma unroll
    for (int r = 0; r < 8; ++r)
      crow[(size_t)(r + 8 * hi) * HID_ + dt * 16 + n] =
          (_Float16)(acc[dt][r] * s_linv[r + 8 * hi]);
}

// ---------------- output projection: ctx @ Wo + bo -> f32 out ----------------
__global__ __launch_bounds__(128)
void k_oproj(const _Float16* __restrict__ ctx, const _Float16* __restrict__ Wot,
             const float* __restrict__ bo, float* __restrict__ out) {
  __shared__ __align__(16) _Float16 lbuf[2][BUFH];

  const int tid  = threadIdx.x;
  const int lane = tid & 31;
  const int wv   = tid >> 5;
  const int n  = lane & 15;
  const int hi = lane >> 4;

  const int rb = blockIdx.x * 128 + wv * 32;   // 64 blocks over B*S rows
  const int n0 = blockIdx.y * 64;              // 16 blocks over HIDDEN cols

  const int cr0 = wv * 16 + ((lane >> 2) & 7);
  const int cb  = (lane & 3) * 16;

  const _Float16* arow0 = ctx + (size_t)(rb + n) * HID_;
  const _Float16* arow1 = arow0 + (size_t)16 * HID_;
  const char* wsrc = (const char*)(Wot + (size_t)(n0 + cr0) * HID_) + cb;
  const uint32_t lds0 =
      (uint32_t)(uintptr_t)&lbuf[0][0] + (uint32_t)cr0 * (LSTR * 2) + cb;

  auto issue = [&](uint32_t dst, int kb) {
    const char* g0 = wsrc + (size_t)kb * 2;
    async_ld_b128(dst, g0);
    async_ld_b128(dst + 8 * (LSTR * 2), g0 + 8 * (HID_ * 2));
  };

  v8f acc[2][4] = {};

  auto compute = [&](const _Float16* lb, int kb) {
    AV a0, a1;
    a0.h[0] = *(const v8h*)(arow0 + kb + 8 * hi);
    a0.h[1] = *(const v8h*)(arow0 + kb + 16 + 8 * hi);
    a1.h[0] = *(const v8h*)(arow1 + kb + 8 * hi);
    a1.h[1] = *(const v8h*)(arow1 + kb + 16 + 8 * hi);
    AV bv[4];
#pragma unroll
    for (int dt = 0; dt < 4; ++dt) {
      const _Float16* bp = lb + (dt * 16 + n) * LSTR + 8 * hi;
      bv[dt].h[0] = *(const v8h*)(bp);
      bv[dt].h[1] = *(const v8h*)(bp + 16);
    }
#pragma unroll
    for (int dt = 0; dt < 4; ++dt) {
      acc[0][dt] = wmma_f16(a0.v, bv[dt].v, acc[0][dt]);
      acc[1][dt] = wmma_f16(a1.v, bv[dt].v, acc[1][dt]);
    }
  };

  issue(lds0, 0);
#pragma unroll 1
  for (int i = 0; i < 31; ++i) {
    issue(lds0 + (uint32_t)(((i & 1) ^ 1) * BUFB), (i + 1) * 32);
    asm volatile("s_wait_asynccnt 2" ::: "memory");
    __syncthreads();
    compute(&lbuf[0][0] + (i & 1) * BUFH, i * 32);
    __syncthreads();
  }
  asm volatile("s_wait_asynccnt 0" ::: "memory");
  __syncthreads();
  compute(&lbuf[1][0], 31 * 32);

#pragma unroll
  for (int dt = 0; dt < 4; ++dt) {
    float bb = bo[n0 + dt * 16 + n];
#pragma unroll
    for (int mi = 0; mi < 2; ++mi)
#pragma unroll
      for (int r = 0; r < 8; ++r)
        out[(size_t)(rb + mi * 16 + r + 8 * hi) * HID_ + n0 + dt * 16 + n] =
            acc[mi][dt][r] + bb;
  }
}

extern "C" void kernel_launch(void* const* d_in, const int* in_sizes, int n_in,
                              void* d_out, int out_size, void* d_ws, size_t ws_size,
                              hipStream_t stream) {
  (void)in_sizes; (void)n_in; (void)out_size; (void)ws_size;
  const float* x  = (const float*)d_in[0];
  const float* Wq = (const float*)d_in[1];
  const float* bq = (const float*)d_in[2];
  const float* Wk = (const float*)d_in[3];
  const float* bk = (const float*)d_in[4];
  const float* Wv = (const float*)d_in[5];
  const float* bv = (const float*)d_in[6];
  const float* Wo = (const float*)d_in[7];
  const float* bo = (const float*)d_in[8];
  float* out = (float*)d_out;

  char* w = (char*)d_ws;
  const size_t SZ_X = (size_t)B_ * S_ * HID_ * 2;   // 16 MB f16
  const size_t SZ_W = (size_t)HID_ * HID_ * 2;      // 2 MB f16
  _Float16* xh  = (_Float16*)w; w += SZ_X;
  _Float16* Wqt = (_Float16*)w; w += SZ_W;
  _Float16* Wkt = (_Float16*)w; w += SZ_W;
  _Float16* Wvt = (_Float16*)w; w += SZ_W;
  _Float16* Wot = (_Float16*)w; w += SZ_W;
  _Float16* Qh  = (_Float16*)w; w += SZ_X;
  _Float16* Kh  = (_Float16*)w; w += SZ_X;
  _Float16* Vt  = (_Float16*)w; w += SZ_X;
  _Float16* ctx = (_Float16*)w; w += SZ_X;

  k_cvt_x <<<8192, 256, 0, stream>>>(x, xh, B_ * S_ * HID_);
  k_cvt_wt<<<4096, 256, 0, stream>>>(Wq, Wqt);
  k_cvt_wt<<<4096, 256, 0, stream>>>(Wk, Wkt);
  k_cvt_wt<<<4096, 256, 0, stream>>>(Wv, Wvt);
  k_cvt_wt<<<4096, 256, 0, stream>>>(Wo, Wot);

  dim3 g1(64, 16);
  k_qkv<<<g1, 128, 0, stream>>>(xh, Wqt, bq, Qh, 0);
  k_qkv<<<g1, 128, 0, stream>>>(xh, Wkt, bk, Kh, 1);
  k_qkv<<<g1, 128, 0, stream>>>(xh, Wvt, bv, Vt, 2);

  k_attn<<<B_ * H_ * (S_ / 16), 32, 0, stream>>>(Qh, Kh, Vt, ctx);

  dim3 g3(64, 16);
  k_oproj<<<g3, 128, 0, stream>>>(ctx, Wot, bo, out);
}
